// CrossAttentionLayer_14173392077304
// MI455X (gfx1250) — matrix operand order repaired and checked by reading, not compiled, not run on hardware
//
#include <hip/hip_runtime.h>
#include <hip/hip_bf16.h>
#include <math.h>

// ---------------------------------------------------------------------------
// CrossAttentionLayer for MI455X (gfx1250, wave32, WMMA bf16 + TDM staging)
//   q = zt @ Wq^T + bq ; k = ic @ Wk^T + bk ; v = ic @ Wv^T + bv
//   out = softmax(q k^T / sqrt(64)) v   (per batch, per head)
// ---------------------------------------------------------------------------

#define B_  4
#define LQ_ 2048
#define LK_ 2048
#define D_  1024
#define H_  16
#define HD_ 64

typedef __bf16 bf16;
typedef __attribute__((ext_vector_type(16))) __bf16 bf16x16;
typedef __attribute__((ext_vector_type(8)))  __bf16 bf16x8;
typedef __attribute__((ext_vector_type(4)))  __bf16 bf16x4;
typedef __attribute__((ext_vector_type(8)))  float  f32x8;
typedef __attribute__((ext_vector_type(4)))  unsigned int u32x4;
typedef __attribute__((ext_vector_type(8)))  int i32x8;
typedef __attribute__((ext_vector_type(4)))  int i32x4;

// D = A*B + C, A 16x32 bf16, B 32x16 bf16, C/D 16x16 f32
__device__ __forceinline__ f32x8 wmma_bf16(bf16x16 a, bf16x16 b, f32x8 c) {
    return __builtin_amdgcn_wmma_f32_16x16x32_bf16(
        /*neg_a=*/false, a, /*neg_b=*/false, b,
        /*c_mod=*/(short)0, c, /*reuse_a=*/false, /*reuse_b=*/false);
}

// A-matrix 16x32 fragment from row-major [M][K] LDS tile, per ISA layout:
// lane = hi*16 + m ; chunks of 8 bf16 at K = hi*8 and K = 16 + hi*8
__device__ __forceinline__ bf16x16 load_a_frag(const bf16* base, int ld, int lane) {
    const int m  = lane & 15;
    const int hi = lane >> 4;
    const bf16* row = base + m * ld;
    union { bf16x16 v; bf16x8 h[2]; } u;
    u.h[0] = *(const bf16x8*)(row + hi * 8);
    u.h[1] = *(const bf16x8*)(row + 16 + hi * 8);
    return u.v;
}

// B-matrix 32x16 fragment from [N][K] LDS tile, per ISA layout:
// lane = hi*16 + n ; chunks of 8 bf16 at K = hi*8 and K = 16 + hi*8
__device__ __forceinline__ bf16x16 load_b_frag(const bf16* base, int ld, int lane) {
    const int n  = lane & 15;
    const int hi = lane >> 4;
    const bf16* row = base + n * ld;
    union { bf16x16 v; bf16x8 h[2]; } u;
    u.h[0] = *(const bf16x8*)(row + hi * 8);
    u.h[1] = *(const bf16x8*)(row + 16 + hi * 8);
    return u.v;
}

// ---------------------------------------------------------------------------
// Tensor Data Mover: DMA a 2-D bf16 tile (rows x row_elems, global row stride
// row_stride_elems) into LDS at lds_byte_off, inserting pad_amount DWORDs of
// LDS padding every pad_interval-coded DWORDs (reproduces padded LDS pitch).
// D# bitfields per CDNA5 ISA 8.3/8.4; higher groups zero => 2-D tensor.
// 6-arg builtin form (clang-23 / therock-10.0 lane).
// ---------------------------------------------------------------------------
__device__ __forceinline__ void tdm_load_2d_bf16(
    const bf16* gsrc, unsigned lds_byte_off,
    unsigned rows, unsigned row_elems, unsigned row_stride_elems,
    unsigned pad_interval_code, unsigned pad_amount_code)
{
    const unsigned long long ga = (unsigned long long)(uintptr_t)gsrc;
    u32x4 g0;
    g0.x = 1u;                                                  // count=1, no gather
    g0.y = lds_byte_off;                                        // lds_addr
    g0.z = (unsigned)(ga & 0xFFFFFFFFu);                        // global_addr[31:0]
    g0.w = (unsigned)((ga >> 32) & 0x01FFFFFFu) | (2u << 30);   // global_addr[56:32] | type=2

    i32x8 g1;
    g1[0] = (int)((1u << 16)                   // data_size = 2 bytes
                | (1u << 20)                   // pad_enable
                | (pad_interval_code << 22)
                | (pad_amount_code << 25));
    g1[1] = (int)((row_elems & 0xFFFFu) << 16);                 // tensor_dim0 lo16
    g1[2] = (int)(((row_elems >> 16) & 0xFFFFu)
                | ((rows & 0xFFFFu) << 16));                    // dim0 hi16 | tensor_dim1 lo16
    g1[3] = (int)(((rows >> 16) & 0xFFFFu)
                | ((row_elems & 0xFFFFu) << 16));               // dim1 hi16 | tile_dim0
    g1[4] = (int)(rows & 0xFFFFu);                              // tile_dim1 | tile_dim2=0
    g1[5] = (int)row_stride_elems;                              // tensor_dim0_stride lo32
    g1[6] = 0;                                                  // stride hi | dim1_stride lo
    g1[7] = 0;

    const i32x4 z4 = {0, 0, 0, 0};
    const i32x8 z8 = {0, 0, 0, 0, 0, 0, 0, 0};
    __builtin_amdgcn_tensor_load_to_lds(g0, g1, z4, z4, z8, 0);
}

// ---------------------------------------------------------------------------
// Kernel 1: f32 -> bf16 conversion (weights)
// ---------------------------------------------------------------------------
__global__ __launch_bounds__(256)
void cvt_f32_bf16_kernel(const float* __restrict__ src, bf16* __restrict__ dst, int n) {
    int i = (blockIdx.x * 256 + threadIdx.x) * 4;
    if (i + 3 < n) {
        float4 f = *(const float4*)(src + i);
        bf16x4 o = { (bf16)f.x, (bf16)f.y, (bf16)f.z, (bf16)f.w };
        *(bf16x4*)(dst + i) = o;
    }
}

// ---------------------------------------------------------------------------
// Kernel 2: projection GEMM  C[M,1024](bf16) = A[M,1024](f32) * W[1024,1024]^T + bias
// WG tile 128x128, 8 waves as 4(M) x 2(N), wave tile 32x64 (2x4 WMMA frags).
// W tile staged by TDM (row 64B = 16 DW -> interval code 3; pad 4 DW -> code 3;
// pitch 80B = 40 bf16). A tile staged manually (needs f32->bf16 convert).
// ---------------------------------------------------------------------------
__global__ __launch_bounds__(256)
void proj_gemm_kernel(const float* __restrict__ A, const bf16* __restrict__ W,
                      const float* __restrict__ bias, bf16* __restrict__ C) {
    __shared__ __align__(16) bf16 ldsA[128][40];   // 32 K + 8 pad
    __shared__ __align__(16) bf16 ldsB[128][40];

    const int tid  = threadIdx.x;
    const int lane = tid & 31;
    const int wave = tid >> 5;
    const int wm   = wave & 3;      // 0..3 -> M subtile of 32
    const int wn   = wave >> 2;     // 0..1 -> N subtile of 64
    const int m0   = blockIdx.y * 128;
    const int n0   = blockIdx.x * 128;

    f32x8 acc[2][4] = {};

    for (int k0 = 0; k0 < D_; k0 += 32) {
        // --- TDM: stage W tile 128x32 bf16 into padded ldsB (wave 0 issues) ---
        if (tid < 32) {
            tdm_load_2d_bf16(W + (size_t)n0 * D_ + k0,
                             (unsigned)(uintptr_t)&ldsB[0][0],
                             /*rows=*/128, /*row_elems=*/32, /*stride=*/D_,
                             /*interval: 16 DW*/3, /*pad: 4 DW*/3);
        }
        // --- stage A tile 128x32 (f32 -> bf16), overlapped with the DMA ---
        {
            const int row = tid >> 1;
            const int c0  = (tid & 1) * 16;
            const float* src = A + (size_t)(m0 + row) * D_ + k0 + c0;
            float4 f0 = *(const float4*)(src + 0);
            float4 f1 = *(const float4*)(src + 4);
            float4 f2 = *(const float4*)(src + 8);
            float4 f3 = *(const float4*)(src + 12);
            bf16x8 o0 = { (bf16)f0.x, (bf16)f0.y, (bf16)f0.z, (bf16)f0.w,
                          (bf16)f1.x, (bf16)f1.y, (bf16)f1.z, (bf16)f1.w };
            bf16x8 o1 = { (bf16)f2.x, (bf16)f2.y, (bf16)f2.z, (bf16)f2.w,
                          (bf16)f3.x, (bf16)f3.y, (bf16)f3.z, (bf16)f3.w };
            *(bf16x8*)&ldsA[row][c0]     = o0;
            *(bf16x8*)&ldsA[row][c0 + 8] = o1;
        }
        if (tid < 32) __builtin_amdgcn_s_wait_tensorcnt(0);
        __syncthreads();

        bf16x16 afrag[2], bfrag[4];
#pragma unroll
        for (int i = 0; i < 2; ++i)
            afrag[i] = load_a_frag(&ldsA[wm * 32 + i * 16][0], 40, lane);
#pragma unroll
        for (int j = 0; j < 4; ++j)
            bfrag[j] = load_b_frag(&ldsB[wn * 64 + j * 16][0], 40, lane);
#pragma unroll
        for (int i = 0; i < 2; ++i)
#pragma unroll
            for (int j = 0; j < 4; ++j)
                acc[i][j] = wmma_bf16(afrag[i], bfrag[j], acc[i][j]);

        __syncthreads();
    }

    // epilogue: one base address per lane, all offsets compile-time constants
    const int hi = lane >> 4, ln = lane & 15;
    bf16* cbase = C + (size_t)(m0 + wm * 32 + hi * 8) * D_ + (n0 + wn * 64 + ln);
    const float* bbase = bias + n0 + wn * 64 + ln;
    float bv[4];
#pragma unroll
    for (int j = 0; j < 4; ++j) bv[j] = bbase[j * 16];
#pragma unroll
    for (int i = 0; i < 2; ++i)
#pragma unroll
        for (int j = 0; j < 4; ++j)
#pragma unroll
            for (int r = 0; r < 8; ++r)
                cbase[(i * 16 + r) * D_ + j * 16] = (bf16)(acc[i][j][r] + bv[j]);
}

// ---------------------------------------------------------------------------
// Kernel 3: flash attention over bf16 Q/K/V, f32 output.
// Grid (LQ/128, H, B); 8 waves; each wave owns 16 query rows; key blocks of 32.
// K tile staged by TDM (row 128B = 32 DW -> interval code 4; pad 4 DW -> code 3;
// pitch 144B = 72 bf16). V transposed in software at staging time.
// ---------------------------------------------------------------------------
__global__ __launch_bounds__(256)
void flash_attn_kernel(const bf16* __restrict__ Qb, const bf16* __restrict__ Kb,
                       const bf16* __restrict__ Vb, float* __restrict__ Out) {
    __shared__ __align__(16) bf16 ldsK[32][72];       // [key][d], pad 8
    __shared__ __align__(16) bf16 ldsV[64][40];       // [d][key] (transposed), pad 8
    __shared__ __align__(16) bf16 ldsP[8][16][40];    // per-wave P staging

    const int tid  = threadIdx.x;
    const int lane = tid & 31;
    const int wave = tid >> 5;
    const int hi   = lane >> 4;
    const int ln   = lane & 15;

    const int b   = blockIdx.z;
    const int h   = blockIdx.y;
    const int q0  = blockIdx.x * 128 + wave * 16;     // this wave's 16 query rows

    const size_t baseQ = (size_t)b * LQ_ * D_ + (size_t)h * HD_;
    const size_t baseK = (size_t)b * LK_ * D_ + (size_t)h * HD_;

    // Q fragments (16 rows x 64 d = 2 A-frags of K=32), straight from global
    bf16x16 qfrag[2];
    {
        const bf16* qrp = Qb + baseQ + (size_t)(q0 + ln) * D_;
#pragma unroll
        for (int f = 0; f < 2; ++f) {
            union { bf16x16 v; bf16x8 h2[2]; } u;
            u.h2[0] = *(const bf16x8*)(qrp + f * 32 + hi * 8);
            u.h2[1] = *(const bf16x8*)(qrp + f * 32 + 16 + hi * 8);
            qfrag[f] = u.v;
        }
    }

    f32x8 o[4] = {};
    float mrun[8], lrun[8];
#pragma unroll
    for (int r = 0; r < 8; ++r) { mrun[r] = -1e30f; lrun[r] = 0.0f; }

    for (int kb = 0; kb < LK_; kb += 32) {
        // --- TDM: stage K tile 32x64 bf16 into padded ldsK (wave 0 issues) ---
        if (tid < 32) {
            tdm_load_2d_bf16(Kb + baseK + (size_t)kb * D_,
                             (unsigned)(uintptr_t)&ldsK[0][0],
                             /*rows=*/32, /*row_elems=*/64, /*stride=*/D_,
                             /*interval: 32 DW*/4, /*pad: 4 DW*/3);
        }
        // --- stage V transposed [64][32] (overlapped with the DMA) ---
        {
            const int key = tid >> 3;
            const int dg  = (tid & 7) * 8;
            const bf16* srcv = Vb + baseK + (size_t)(kb + key) * D_ + dg;
            bf16x8 vv = *(const bf16x8*)srcv;
#pragma unroll
            for (int i = 0; i < 8; ++i) ldsV[dg + i][key] = vv[i];
        }
        if (tid < 32) __builtin_amdgcn_s_wait_tensorcnt(0);
        __syncthreads();

        // --- S = Q K^T * scale : two 16x16 tiles (j), each K=64 over 2 chunks
        f32x8 s[2];
#pragma unroll
        for (int j = 0; j < 2; ++j) {
            f32x8 c = {};
#pragma unroll
            for (int dc = 0; dc < 2; ++dc) {
                bf16x16 kf = load_b_frag(&ldsK[j * 16][dc * 32], 72, lane);
                c = wmma_bf16(qfrag[dc], kf, c);
            }
            s[j] = c * 0.125f;   // 1/sqrt(64)
        }

        // --- online softmax (row = r + 8*hi ; cols spread over 16-lane half)
        float alpha[8], p0[8], p1[8];
#pragma unroll
        for (int r = 0; r < 8; ++r) {
            float mx = fmaxf(s[0][r], s[1][r]);
#pragma unroll
            for (int off = 1; off < 16; off <<= 1)
                mx = fmaxf(mx, __shfl_xor(mx, off, 32));
            const float mnew = fmaxf(mrun[r], mx);
            alpha[r] = __expf(mrun[r] - mnew);
            p0[r] = __expf(s[0][r] - mnew);
            p1[r] = __expf(s[1][r] - mnew);
            float sum = p0[r] + p1[r];
#pragma unroll
            for (int off = 1; off < 16; off <<= 1)
                sum += __shfl_xor(sum, off, 32);
            lrun[r] = lrun[r] * alpha[r] + sum;
            mrun[r] = mnew;
        }
#pragma unroll
        for (int c = 0; c < 4; ++c)
#pragma unroll
            for (int r = 0; r < 8; ++r)
                o[c][r] *= alpha[r];

        // --- restage P (C layout -> A layout) through per-wave LDS ---
#pragma unroll
        for (int r = 0; r < 8; ++r) {
            ldsP[wave][r + hi * 8][ln]      = (bf16)p0[r];
            ldsP[wave][r + hi * 8][16 + ln] = (bf16)p1[r];
        }
        bf16x16 pf = load_a_frag(&ldsP[wave][0][0], 40, lane);

        // --- O += P V : 4 d-chunks of 16 cols, K = 32 keys ---
#pragma unroll
        for (int c = 0; c < 4; ++c) {
            bf16x16 vf = load_b_frag(&ldsV[c * 16][0], 40, lane);
            o[c] = wmma_bf16(pf, vf, o[c]);
        }
        __syncthreads();
    }

    // --- epilogue: out[b, q, h*64 + d] = O / l ; constant store offsets ---
    float* obase = Out + ((size_t)b * LQ_ + q0 + hi * 8) * D_ + h * HD_ + ln;
#pragma unroll
    for (int r = 0; r < 8; ++r) {
        const float inv = 1.0f / lrun[r];
#pragma unroll
        for (int c = 0; c < 4; ++c)
            obase[r * D_ + c * 16] = o[c][r] * inv;
    }
}

// ---------------------------------------------------------------------------
extern "C" void kernel_launch(void* const* d_in, const int* in_sizes, int n_in,
                              void* d_out, int out_size, void* d_ws, size_t ws_size,
                              hipStream_t stream) {
    (void)in_sizes; (void)n_in; (void)out_size; (void)ws_size;

    const float* zt = (const float*)d_in[0];
    const float* ic = (const float*)d_in[1];
    const float* Wq = (const float*)d_in[2];
    const float* bq = (const float*)d_in[3];
    const float* Wk = (const float*)d_in[4];
    const float* bk = (const float*)d_in[5];
    const float* Wv = (const float*)d_in[6];
    const float* bv = (const float*)d_in[7];
    float* out = (float*)d_out;

    char* ws = (char*)d_ws;
    const size_t actBytes = (size_t)B_ * LQ_ * D_ * sizeof(bf16);   // 16 MB each
    const size_t wBytes   = (size_t)D_ * D_ * sizeof(bf16);         // 2 MB each
    bf16* qb  = (bf16*)(ws);
    bf16* kb  = (bf16*)(ws + actBytes);
    bf16* vb  = (bf16*)(ws + 2 * actBytes);
    bf16* Wqb = (bf16*)(ws + 3 * actBytes);
    bf16* Wkb = (bf16*)(ws + 3 * actBytes + wBytes);
    bf16* Wvb = (bf16*)(ws + 3 * actBytes + 2 * wBytes);

    // 1) weights f32 -> bf16
    {
        const int n = D_ * D_;
        dim3 g(n / (256 * 4));
        cvt_f32_bf16_kernel<<<g, 256, 0, stream>>>(Wq, Wqb, n);
        cvt_f32_bf16_kernel<<<g, 256, 0, stream>>>(Wk, Wkb, n);
        cvt_f32_bf16_kernel<<<g, 256, 0, stream>>>(Wv, Wvb, n);
    }
    // 2) projections (M = B*L = 8192, N = K = 1024)
    {
        dim3 grid(D_ / 128, (B_ * LQ_) / 128);
        proj_gemm_kernel<<<grid, 256, 0, stream>>>(zt, Wqb, bq, qb);
        proj_gemm_kernel<<<grid, 256, 0, stream>>>(ic, Wkb, bk, kb);
        proj_gemm_kernel<<<grid, 256, 0, stream>>>(ic, Wvb, bv, vb);
    }
    // 3) flash attention
    {
        dim3 grid(LQ_ / 128, H_, B_);
        flash_attn_kernel<<<grid, 256, 0, stream>>>(qb, kb, vb, out);
    }
}
